// Mask_MHCrossAttention_87668872446303
// MI455X (gfx1250) — matrix operand order repaired and checked by reading, not compiled, not run on hardware
//
#include <hip/hip_runtime.h>

// ---------------------------------------------------------------------------
// Masked multi-head cross-attention for MI455X (gfx1250, wave32, WMMA bf16).
// B=8, S=1024, H=16, DQ=DH=64, LAT=DV=1024.
// Roofline: mask stream 512MB (~22us @23.3TB/s) vs ~103 GFLOP (~20us @ bf16
// WMMA dense peak) -> both matter. GEMMs: 64x64 wave tiles, 16 WMMAs per
// 32-K step; A and W packed so every fragment loads as global_load_b128
// (W/V packed COLUMN-major so B-fragment dwords are contiguous per lane).
// Attention is mask-bound: 16 q-rows/wave, mask prefetch, online softmax.
// ---------------------------------------------------------------------------

#define BB   8
#define SS   1024
#define HH   16
#define DD   64
#define LATD 1024   // = HH*DD
#define KP2  (LATD / 2)   // packed K pairs per row/col
#define SP2  (SS / 2)     // packed key pairs

typedef __attribute__((ext_vector_type(16))) __bf16        v16bf;
typedef __attribute__((ext_vector_type(8)))  float         v8f;
typedef __attribute__((ext_vector_type(8)))  unsigned int  v8u;

// f32 pair -> packed bf16 dword (lo in [15:0]); RNE (HW cvt if available)
static __device__ __forceinline__ unsigned pk_bf16(float lo, float hi) {
#if __has_builtin(__builtin_amdgcn_cvt_pk_bf16_f32)
    typedef __attribute__((ext_vector_type(2))) __bf16 v2bf;
    v2bf p = __builtin_amdgcn_cvt_pk_bf16_f32(lo, hi);
    return __builtin_bit_cast(unsigned, p);
#else
    unsigned a = __float_as_uint(lo), b = __float_as_uint(hi);
    a += 0x7FFFu + ((a >> 16) & 1u);
    b += 0x7FFFu + ((b >> 16) & 1u);
    return (a >> 16) | (b & 0xFFFF0000u);
#endif
}
static __device__ __forceinline__ unsigned short bf16s(float x) {
    return (unsigned short)(pk_bf16(x, 0.f) & 0xFFFFu);
}

static __device__ __forceinline__ v8f wmma_bf16(v8u a, v8u b, v8f c) {
    return __builtin_amdgcn_wmma_f32_16x16x32_bf16(
        /*neg_a=*/false, __builtin_bit_cast(v16bf, a),
        /*neg_b=*/false, __builtin_bit_cast(v16bf, b),
        /*c_mod=*/(short)0, c, /*reuse_a=*/false, /*reuse_b=*/false);
}

// ---------------------------------------------------------------------------
// Weight pre-pack, COLUMN-major: Wp[n * KP2 + k/2] = (W[k,n], W[k+1,n]).
// A lane's B-fragment (8 consecutive k-pairs of one column) is contiguous
// -> 2x global_load_b128 per fragment in the GEMMs.
// ---------------------------------------------------------------------------
__global__ __launch_bounds__(256) void pack_w_kernel(
    const float* __restrict__ W, unsigned* __restrict__ Wp)
{
    const int i  = blockIdx.x * 256 + threadIdx.x;   // pair idx, 512*1024 total
    const int k2 = i >> 10;                          // coalesced reads
    const int n  = i & (LATD - 1);
    Wp[(size_t)n * KP2 + k2] =
        pk_bf16(W[(size_t)(2 * k2) * LATD + n], W[(size_t)(2 * k2 + 1) * LATD + n]);
}

// ---------------------------------------------------------------------------
// Activation pre-pack (pairs contiguous along K): dst[i] = (src[2i], src[2i+1])
// ---------------------------------------------------------------------------
__global__ __launch_bounds__(256) void pack_x_kernel(
    const float* __restrict__ src, unsigned* __restrict__ dst)
{
    const size_t i = (size_t)blockIdx.x * 256 + threadIdx.x;  // 4M dwords
    dst[i] = pk_bf16(src[2 * i], src[2 * i + 1]);
}

// ---------------------------------------------------------------------------
// GEMM core: 64x64 wave tile, A packed row-major (pairs along K), W packed
// column-major. 16 WMMAs per 32-K step; all fragment loads b128-mergeable.
// MODE: 1 = bf16 head-major [B,H,S,64] (Q,K); 2 = bf16 V layout
// [bh][d][keypair*2+parity]; 3 = f32 row-major (final output).
// grid = 256 blocks x 8 waves = 2048 wave tiles (128 x 16).
// ---------------------------------------------------------------------------
template <int MODE, typename OutT>
__global__ __launch_bounds__(256) void gemm_kernel(
    const unsigned* __restrict__ Apk, const unsigned* __restrict__ Wp,
    const float* __restrict__ bias, OutT* __restrict__ outp)
{
    const int wave = threadIdx.x >> 5, lane = threadIdx.x & 31;
    const int lh = lane & 15, hi = lane >> 4;
    const int t  = blockIdx.x * 8 + wave;
    const int tn = t & 15, tm = t >> 4;
    const int m0 = tm * 64, n0 = tn * 64;

    v8f c[4][4];
    #pragma unroll
    for (int ns = 0; ns < 4; ++ns) {
        const float bv = bias[n0 + ns * 16 + lh];
        #pragma unroll
        for (int ms = 0; ms < 4; ++ms)
            #pragma unroll
            for (int r = 0; r < 8; ++r) c[ms][ns][r] = bv;
    }

    const int cbA2 = hi ? 8 : 0;    // A-frag K sub-offset (pair units)
    const int kbB2 = hi ? 8 : 0;    // B-frag K sub-offset (pair units)
    const unsigned* Arow[4];
    const unsigned* Wcol[4];
    #pragma unroll
    for (int ms = 0; ms < 4; ++ms)
        Arow[ms] = Apk + (size_t)(m0 + ms * 16 + lh) * KP2 + cbA2;
    #pragma unroll
    for (int ns = 0; ns < 4; ++ns)
        Wcol[ns] = Wp + (size_t)(n0 + ns * 16 + lh) * KP2 + kbB2;

    for (int k2 = 0; k2 < KP2; k2 += 16) {          // 32 K elements per step
        v8u a[4], b[4];
        #pragma unroll
        for (int ms = 0; ms < 4; ++ms)
            #pragma unroll
            for (int v = 0; v < 4; ++v) {
                a[ms][v]     = Arow[ms][k2 + v];
                a[ms][4 + v] = Arow[ms][k2 + 4 + v];
            }
        #pragma unroll
        for (int ns = 0; ns < 4; ++ns)
            #pragma unroll
            for (int v = 0; v < 8; ++v)
                b[ns][v] = Wcol[ns][k2 + v];
        #pragma unroll
        for (int ns = 0; ns < 4; ++ns)
            #pragma unroll
            for (int ms = 0; ms < 4; ++ms)
                c[ms][ns] = wmma_bf16(a[ms], b[ns], c[ms][ns]);
    }

    const int b_ = m0 / SS, s_ = m0 % SS, h_ = n0 / DD;
    const size_t bh = (size_t)(b_ * HH + h_);
    #pragma unroll
    for (int ms = 0; ms < 4; ++ms) {
        #pragma unroll
        for (int ns = 0; ns < 4; ++ns) {
            #pragma unroll
            for (int r = 0; r < 8; ++r) {
                const int row = s_ + ms * 16 + r + hi * 8;
                const int d   = ns * 16 + lh;
                if (MODE == 1) {        // head-major [B,H,S,DD] bf16
                    outp[(bh * SS + row) * DD + d] = (OutT)bf16s(c[ms][ns][r]);
                } else if (MODE == 2) { // V: [bh][d][keypair], parity-interleaved
                    outp[((bh * DD + d) * SP2 + (row >> 1)) * 2 + (row & 1)] =
                        (OutT)bf16s(c[ms][ns][r]);
                } else {                // f32 row-major [M, N]
                    outp[(size_t)(m0 + ms * 16 + r + hi * 8) * LATD +
                         (n0 + ns * 16 + lh)] = (OutT)c[ms][ns][r];
                }
            }
        }
    }
}

// ---------------------------------------------------------------------------
// Flash attention, one wave = 16 query rows of one (b,h); 32-key blocks.
// Mask-HBM-bound: small tiles for occupancy + mask prefetch. C-fragment row
// split (lanes 0-15: M=r, lanes 16-31: M=8+r) -> softmax row stats reduce
// over 16-lane halves with shfl_xor {8,4,2,1}.
// ---------------------------------------------------------------------------
__global__ __launch_bounds__(256) void attn_kernel(
    const unsigned short* __restrict__ Qbf, const unsigned short* __restrict__ Kbf,
    const unsigned* __restrict__ Vp, const int* __restrict__ mask,
    unsigned short* __restrict__ attn)
{
    __shared__ unsigned short Pb[8][16 * 32];   // per-wave P tile (bf16)

    const int wave = threadIdx.x >> 5, lane = threadIdx.x & 31;
    const int lh = lane & 15, hi = lane >> 4;
    const int tile = blockIdx.x;                 // 1024 = B*H * (S/128)
    const int qblk = tile & 7;
    const int bh   = tile >> 3;                  // b*H + h
    const int b_ = bh >> 4, h_ = bh & 15;
    const int q0 = qblk * 128 + wave * 16;

    const unsigned short* Qh = Qbf + (size_t)bh * SS * DD;
    const unsigned short* Kh = Kbf + (size_t)bh * SS * DD;
    const unsigned*       Vh = Vp  + (size_t)bh * DD * SP2;
    const int* Mh = mask + (size_t)bh * SS * SS;
    unsigned short* Pw = &Pb[wave][0];

    const int cbA = hi ? 16 : 0, kbB2 = hi ? 8 : 0;

    // Q A-fragments (16x64 = two 16x32), contiguous bf16 pairs -> b128 loads
    v8u aq[2];
    const unsigned short* Qrow = Qh + (size_t)(q0 + lh) * DD;
    #pragma unroll
    for (int tt = 0; tt < 2; ++tt) {
        #pragma unroll
        for (int v = 0; v < 4; ++v) {
            aq[tt][v]     = *(const unsigned*)(const void*)(Qrow + tt*32 + cbA + 2*v);
            aq[tt][4 + v] = *(const unsigned*)(const void*)(Qrow + tt*32 + cbA + 8 + 2*v);
        }
    }

    v8f acc[4];
    #pragma unroll
    for (int i = 0; i < 4; ++i)
        #pragma unroll
        for (int r = 0; r < 8; ++r) acc[i][r] = 0.f;
    float m_[8], l_[8];
    #pragma unroll
    for (int r = 0; r < 8; ++r) { m_[r] = -5.0e29f; l_[r] = 0.f; }

    for (int j = 0; j < SS; j += 32) {
        // prefetch next mask block (the 512MB streamed array)
        if (j + 32 < SS)
            __builtin_prefetch(Mh + (size_t)(q0 + lh) * SS + (j + 32) + hi * 16, 0, 0);

        // ---- scores: two 16x16 tiles = Q(16x64) x K^T(64x32) -------------
        v8f s[2];
        #pragma unroll
        for (int n = 0; n < 2; ++n) {
            const int key = j + n * 16 + lh;
            const unsigned short* Krow = Kh + (size_t)key * DD;   // K^T col
            v8u b0, b1;
            #pragma unroll
            for (int v = 0; v < 8; ++v) {
                b0[v] = *(const unsigned*)(const void*)(Krow + 2*kbB2 + 2*v);      // d 0..31
                b1[v] = *(const unsigned*)(const void*)(Krow + 32 + 2*kbB2 + 2*v); // d 32..63
            }
            v8f cz;
            #pragma unroll
            for (int r = 0; r < 8; ++r) cz[r] = 0.f;
            cz = wmma_bf16(aq[0], b0, cz);
            cz = wmma_bf16(aq[1], b1, cz);
            // scale + mask (mask==0 -> -inf surrogate)
            #pragma unroll
            for (int r = 0; r < 8; ++r) {
                const int mrow = q0 + r + hi * 8;
                const int mv = Mh[(size_t)mrow * SS + key];
                cz[r] = (mv == 0) ? -1.0e30f : cz[r] * 0.125f;
            }
            s[n] = cz;
        }

        // ---- online softmax over the 32 new columns ----------------------
        #pragma unroll
        for (int r = 0; r < 8; ++r) {
            float x = fmaxf(s[0][r], s[1][r]);
            #pragma unroll
            for (int off = 8; off >= 1; off >>= 1)
                x = fmaxf(x, __shfl_xor(x, off, 32));
            const float mn = fmaxf(m_[r], x);
            const float al = __expf(m_[r] - mn);
            const float p0 = __expf(s[0][r] - mn);
            const float p1 = __expf(s[1][r] - mn);
            s[0][r] = p0; s[1][r] = p1;
            float rs = p0 + p1;
            #pragma unroll
            for (int off = 8; off >= 1; off >>= 1)
                rs += __shfl_xor(rs, off, 32);
            l_[r] = l_[r] * al + rs;
            m_[r] = mn;
            #pragma unroll
            for (int tt = 0; tt < 4; ++tt) acc[tt][r] *= al;
        }

        // ---- C-layout P -> LDS -> A-layout P (16x32 bf16, wave private) --
        #pragma unroll
        for (int n = 0; n < 2; ++n)
            #pragma unroll
            for (int r = 0; r < 8; ++r)
                Pw[(r + hi * 8) * 32 + n * 16 + lh] = bf16s(s[n][r]);

        v8u ap;
        #pragma unroll
        for (int v = 0; v < 4; ++v) {
            ap[v]     = *(const unsigned*)(const void*)(&Pw[lh * 32 + cbA + 2*v]);
            ap[4 + v] = *(const unsigned*)(const void*)(&Pw[lh * 32 + cbA + 8 + 2*v]);
        }

        // ---- acc += P(16x32) x V(32x64): V column-packed -> b128 loads ---
        const int jp = (j >> 1) + kbB2;
        #pragma unroll
        for (int tt = 0; tt < 4; ++tt) {
            const unsigned* Vcol = Vh + (size_t)(tt * 16 + lh) * SP2 + jp;
            v8u bv;
            #pragma unroll
            for (int v = 0; v < 8; ++v)
                bv[v] = Vcol[v];
            acc[tt] = wmma_bf16(ap, bv, acc[tt]);
        }
    }

    // ---- finalize: /l, store bf16 row-major [B*S, H*DD] ------------------
    #pragma unroll
    for (int r = 0; r < 8; ++r) {
        const float inv = (l_[r] > 0.f) ? 1.f / l_[r] : 0.f;
        const int row = q0 + r + hi * 8;
        #pragma unroll
        for (int tt = 0; tt < 4; ++tt) {
            const int col = h_ * DD + tt * 16 + lh;
            attn[((size_t)b_ * SS + row) * LATD + col] = bf16s(acc[tt][r] * inv);
        }
    }
}

// ---------------------------------------------------------------------------
extern "C" void kernel_launch(void* const* d_in, const int* in_sizes, int n_in,
                              void* d_out, int out_size, void* d_ws, size_t ws_size,
                              hipStream_t stream)
{
    const float* input  = (const float*)d_in[0];
    const float* latent = (const float*)d_in[1];
    const int*   mask   = (const int*)d_in[2];
    const float* Wq = (const float*)d_in[3];
    const float* bq = (const float*)d_in[4];
    const float* Wk = (const float*)d_in[5];
    const float* bk = (const float*)d_in[6];
    const float* Wv = (const float*)d_in[7];
    const float* bv = (const float*)d_in[8];
    const float* Wo = (const float*)d_in[9];
    const float* bo = (const float*)d_in[10];

    // workspace: Q|K|V|attn (64MB) | 4x packed W (8MB) | packed input+latent
    // (32MB) = 104MB total
    const size_t qkv  = (size_t)BB * HH * SS * DD;       // 8M bf16 elements
    const size_t wpel = (size_t)KP2 * LATD;              // 512K dwords per W
    const size_t xpel = (size_t)BB * SS * KP2;           // 4M dwords per act
    unsigned short* wsQ = (unsigned short*)d_ws;
    unsigned short* wsK = wsQ + qkv;
    unsigned short* wsV = wsK + qkv;
    unsigned short* wsA = wsV + qkv;
    unsigned* wpQ = (unsigned*)(wsA + qkv);
    unsigned* wpK = wpQ + wpel;
    unsigned* wpV = wpK + wpel;
    unsigned* wpO = wpV + wpel;
    unsigned* xin = wpO + wpel;     // packed input  (bf16 pairs)
    unsigned* xla = xin + xpel;     // packed latent (bf16 pairs)

    dim3 blk(256);
    pack_w_kernel<<<2048,  blk, 0, stream>>>(Wq, wpQ);
    pack_w_kernel<<<2048,  blk, 0, stream>>>(Wk, wpK);
    pack_w_kernel<<<2048,  blk, 0, stream>>>(Wv, wpV);
    pack_w_kernel<<<2048,  blk, 0, stream>>>(Wo, wpO);
    pack_x_kernel<<<16384, blk, 0, stream>>>(input,  xin);
    pack_x_kernel<<<16384, blk, 0, stream>>>(latent, xla);

    gemm_kernel<1, unsigned short><<<256, blk, 0, stream>>>(xla, wpQ, bq, wsQ);
    gemm_kernel<1, unsigned short><<<256, blk, 0, stream>>>(xin, wpK, bk, wsK);
    gemm_kernel<2, unsigned short><<<256, blk, 0, stream>>>(xin, wpV, bv, wsV);
    attn_kernel<<<1024, blk, 0, stream>>>(wsQ, wsK, (const unsigned*)wsV, mask, wsA);
    gemm_kernel<3, float><<<256, blk, 0, stream>>>((const unsigned*)wsA, wpO, bo,
                                                   (float*)d_out);
}
